// MoE_2723009265966
// MI455X (gfx1250) — matrix operand order repaired and checked by reading, not compiled, not run on hardware
//
#include <hip/hip_runtime.h>
#include <math.h>

// ---------------- problem constants (from reference) ----------------
#define NTOK 4096      // B*T
#define DDIM 1024
#define HDIM 4096
#define NEXP 8
#define CAP  1280      // int(1.25 * 4096 * 2 / 8)

// ---------------- types ----------------
typedef __attribute__((ext_vector_type(8)))  __bf16 bf16x8;
typedef __attribute__((ext_vector_type(16))) __bf16 bf16x16;
typedef __attribute__((ext_vector_type(8)))  float  v8f;
typedef __attribute__((ext_vector_type(4)))  int    v4i;

union BV { bf16x16 v; bf16x8 h[2]; };

__device__ inline unsigned short f2bf_rn(float f) {
    unsigned u = __float_as_uint(f);
    u += 0x7FFFu + ((u >> 16) & 1u);
    return (unsigned short)(u >> 16);
}
__device__ inline float bf2f(unsigned short h) {
    return __uint_as_float(((unsigned)h) << 16);
}
__device__ inline unsigned pack2(unsigned short a, unsigned short b) {
    return (unsigned)a | ((unsigned)b << 16);
}
// split f32 -> (hi, lo) bf16
__device__ inline void split_bf16(float f, unsigned short& hi, unsigned short& lo) {
    hi = f2bf_rn(f);
    lo = f2bf_rn(f - bf2f(hi));
}

// ---- CDNA5 async global->LDS copy (16B per lane), with fallback ----
__device__ inline void async_cp16(void* lds, const void* g) {
#if defined(__AMDGCN__) && __has_builtin(__builtin_amdgcn_global_load_async_to_lds_b128)
    __builtin_amdgcn_global_load_async_to_lds_b128(
        (__attribute__((address_space(1))) v4i*)(g),
        (__attribute__((address_space(3))) v4i*)(lds), 0, 0);
#else
    *(uint4*)lds = *(const uint4*)g;
#endif
}
__device__ inline void wait_async0() {
#if defined(__AMDGCN__) && __has_builtin(__builtin_amdgcn_s_wait_asynccnt)
    __builtin_amdgcn_s_wait_asynccnt(0);
#elif defined(__AMDGCN__)
    asm volatile("s_wait_asynccnt 0x0" ::: "memory");
#endif
}

// =====================================================================
// 1) Router: logits = x @ gate_w ; top-2 ; softmax over the 2 logits
// =====================================================================
__global__ void router_kernel(const float* __restrict__ x,
                              const float* __restrict__ gate_w,
                              int* __restrict__ top_idx,     // [NTOK][2]
                              float* __restrict__ top_p)     // [NTOK][2]
{
    const int wave = threadIdx.x >> 5;
    const int lane = threadIdx.x & 31;
    const int t = blockIdx.x * 8 + wave;
    if (t >= NTOK) return;

    float acc[8];
#pragma unroll
    for (int e = 0; e < 8; ++e) acc[e] = 0.0f;

    for (int d = lane; d < DDIM; d += 32) {
        float xv = x[(size_t)t * DDIM + d];
        const float4 g0 = *(const float4*)(gate_w + (size_t)d * NEXP);
        const float4 g1 = *(const float4*)(gate_w + (size_t)d * NEXP + 4);
        acc[0] += xv * g0.x; acc[1] += xv * g0.y;
        acc[2] += xv * g0.z; acc[3] += xv * g0.w;
        acc[4] += xv * g1.x; acc[5] += xv * g1.y;
        acc[6] += xv * g1.z; acc[7] += xv * g1.w;
    }
#pragma unroll
    for (int off = 16; off > 0; off >>= 1) {
#pragma unroll
        for (int e = 0; e < 8; ++e) acc[e] += __shfl_xor(acc[e], off, 32);
    }
    if (lane == 0) {
        int i0 = 0; float l0 = acc[0];
#pragma unroll
        for (int e = 1; e < 8; ++e) if (acc[e] > l0) { l0 = acc[e]; i0 = e; }
        int i1 = -1; float l1 = -3.4e38f;
#pragma unroll
        for (int e = 0; e < 8; ++e) if (e != i0 && acc[e] > l1) { l1 = acc[e]; i1 = e; }
        float p0 = 1.0f / (1.0f + expf(l1 - l0));
        top_idx[t * 2 + 0] = i0;  top_idx[t * 2 + 1] = i1;
        top_p [t * 2 + 0] = p0;  top_p [t * 2 + 1] = 1.0f - p0;
    }
}

// =====================================================================
// 2) Deterministic capacity assignment (matches reference cumsum order)
// =====================================================================
__global__ void assign_kernel(const int* __restrict__ top_idx,
                              int* __restrict__ dispatch_tok,   // [E][CAP]
                              int* __restrict__ combine_slot,   // [NTOK][2]
                              int* __restrict__ used_count)     // [E]
{
    const int e    = blockIdx.x;
    const int tid  = threadIdx.x;
    const int lane = tid & 31;
    const int wave = tid >> 5;
    __shared__ int wsum[8];

    int running = 0;
    const int total = 2 * NTOK;
    for (int base = 0; base < total; base += 256) {
        int i = base + tid;
        int k = i / NTOK;
        int t = i - k * NTOK;
        int flag = (top_idx[t * 2 + k] == e) ? 1 : 0;

        unsigned m = (unsigned)__ballot(flag);
        int prefix = __popc(m & ((1u << lane) - 1u));
        if (lane == 0) wsum[wave] = __popc(m);
        __syncthreads();
        int woff = 0;
        for (int w2 = 0; w2 < wave; ++w2) woff += wsum[w2];
        int btotal = 0;
        for (int w2 = 0; w2 < 8; ++w2) btotal += wsum[w2];
        if (flag) {
            int rank = running + woff + prefix;
            if (rank < CAP) {
                dispatch_tok[e * CAP + rank] = t;
                combine_slot[t * 2 + k] = rank;
            } else {
                combine_slot[t * 2 + k] = -1;
            }
        }
        running += btotal;
        __syncthreads();
    }
    if (tid == 0) used_count[e] = running < CAP ? running : CAP;
}

// =====================================================================
// 3) Dispatch gather + split: xb_hi/xb_lo[e][slot][:] = split(x[tok])
// =====================================================================
__global__ void dispatch_kernel(const float* __restrict__ x,
                                const int* __restrict__ dispatch_tok,
                                const int* __restrict__ used_count,
                                unsigned short* __restrict__ xb_hi,  // [E][CAP][D]
                                unsigned short* __restrict__ xb_lo)
{
    const int s = blockIdx.x;
    const int e = blockIdx.y;
    const int cnt = used_count[e];
    float4 v = make_float4(0.f, 0.f, 0.f, 0.f);
    if (s < cnt) {
        int tok = dispatch_tok[e * CAP + s];
        v = ((const float4*)x)[(size_t)tok * (DDIM / 4) + threadIdx.x];
    }
    unsigned short hx, hy, hz, hw, lx, ly, lz, lw;
    split_bf16(v.x, hx, lx); split_bf16(v.y, hy, ly);
    split_bf16(v.z, hz, lz); split_bf16(v.w, hw, lw);
    size_t idx = ((size_t)e * CAP + s) * (DDIM / 4) + threadIdx.x; // uint2 units
    ((uint2*)xb_hi)[idx] = make_uint2(pack2(hx, hy), pack2(hz, hw));
    ((uint2*)xb_lo)[idx] = make_uint2(pack2(lx, ly), pack2(lz, lw));
}

// =====================================================================
// 4) Split-bf16 WMMA GEMM:  C[M,N] = act(A[M,K] @ B[K,N] + bias[N])
//    A supplied as bf16 hi/lo planes -> async DMA straight into LDS.
//    B (weights, fp32) staged + split + transposed to LDS [N][K].
//    3 WMMA terms: Ahi*Bhi + Ahi*Blo + Alo*Bhi, fp32 accumulate.
//    FFN1: +bias, exact GELU, store bf16 hi/lo planes (input of FFN2)
//    FFN2: +bias, store fp32
// =====================================================================
#define LDT 40   // LDS row stride in bf16 elements (32 + 8 pad)

template <bool FFN1>
__global__ __launch_bounds__(256)
void moe_gemm(const unsigned short* __restrict__ Ahi,  // [E][M][K] bf16
              const unsigned short* __restrict__ Alo,
              const float* __restrict__ B,             // [E][K][N] fp32
              const float* __restrict__ bias,          // [E][N]
              unsigned short* __restrict__ Chi,        // FFN1 out planes
              unsigned short* __restrict__ Clo,
              float* __restrict__ Cf,                  // FFN2 out
              int M, int N, int K)
{
    const size_t eo = blockIdx.z;
    const unsigned short* AhiE = Ahi + eo * (size_t)M * K;
    const unsigned short* AloE = Alo + eo * (size_t)M * K;
    const float* Be = B + eo * (size_t)K * N;
    const float* be = bias + eo * (size_t)N;

    const int m0 = blockIdx.y * 128;
    const int n0 = blockIdx.x * 128;
    const int tid = threadIdx.x;
    const int lane = tid & 31, wave = tid >> 5;
    const int wm = wave >> 1, wn = wave & 1;
    const int half = lane >> 4, l16 = lane & 15;

    __shared__ __align__(16) unsigned short AsHi[128 * LDT];
    __shared__ __align__(16) unsigned short AsLo[128 * LDT];
    __shared__ __align__(16) unsigned short BsHi[128 * LDT];
    __shared__ __align__(16) unsigned short BsLo[128 * LDT];

    v8f acc[2][4];
#pragma unroll
    for (int i = 0; i < 2; ++i)
#pragma unroll
        for (int j = 0; j < 4; ++j)
            acc[i][j] = (v8f){0.f, 0.f, 0.f, 0.f, 0.f, 0.f, 0.f, 0.f};

    for (int kt = 0; kt < K; kt += 32) {
        // ---- A tiles: async DMA bf16 planes -> LDS (ASYNCcnt) ----
        // tile = 128 rows x 32 bf16 (64B) = 4 x 16B chunks per row
#pragma unroll
        for (int it = 0; it < 2; ++it) {
            int c   = tid + it * 256;    // 0..511
            int row = c >> 2;
            int ch  = c & 3;
            size_t goff = (size_t)(m0 + row) * K + kt + ch * 8;
            int    loff = row * LDT + ch * 8;
            async_cp16(&AsHi[loff], AhiE + goff);
            async_cp16(&AsLo[loff], AloE + goff);
        }
        // ---- B tile 32x128 fp32: coalesced column loads, split, packed
        //      transposed store to LDS [N][K] (overlaps with A DMA) ----
#pragma unroll
        for (int it = 0; it < 4; ++it) {
            int q    = tid + it * 256;   // 0..1023
            int nloc = q & 127;
            int k4   = q >> 7;           // 0..7 (4 K each)
            const float* bp = Be + (size_t)(kt + k4 * 4) * N + n0 + nloc;
            float f0 = bp[0];
            float f1 = bp[(size_t)N];
            float f2 = bp[(size_t)2 * N];
            float f3 = bp[(size_t)3 * N];
            unsigned short h0, h1, h2, h3, l0, l1, l2, l3;
            split_bf16(f0, h0, l0); split_bf16(f1, h1, l1);
            split_bf16(f2, h2, l2); split_bf16(f3, h3, l3);
            int idx = nloc * LDT + k4 * 4;               // 8-byte aligned
            *(uint2*)&BsHi[idx] = make_uint2(pack2(h0, h1), pack2(h2, h3));
            *(uint2*)&BsLo[idx] = make_uint2(pack2(l0, l1), pack2(l2, l3));
        }
        wait_async0();
        __syncthreads();

        // ---- fragments (ISA 7.12.2 layouts) + 24 WMMAs ----
        BV aHi[2], aLo[2];
#pragma unroll
        for (int fm = 0; fm < 2; ++fm) {
            int mr = wm * 32 + fm * 16 + l16;
            const unsigned short* p = &AsHi[mr * LDT + half * 8];
            aHi[fm].h[0] = *(const bf16x8*)(p);        // K: 8*half + 0..7
            aHi[fm].h[1] = *(const bf16x8*)(p + 16);   // K: 16 + 8*half + 0..7
            const unsigned short* q = &AsLo[mr * LDT + half * 8];
            aLo[fm].h[0] = *(const bf16x8*)(q);
            aLo[fm].h[1] = *(const bf16x8*)(q + 16);
        }
#pragma unroll
        for (int fn = 0; fn < 4; ++fn) {
            int nc = wn * 64 + fn * 16 + l16;
            BV bHi, bLo;
            const unsigned short* p = &BsHi[nc * LDT + half * 8];
            bHi.h[0] = *(const bf16x8*)(p);            // K: 8*half + 0..7
            bHi.h[1] = *(const bf16x8*)(p + 16);       // K: 16 + 8*half + 0..7
            const unsigned short* q = &BsLo[nc * LDT + half * 8];
            bLo.h[0] = *(const bf16x8*)(q);
            bLo.h[1] = *(const bf16x8*)(q + 16);
#pragma unroll
            for (int fm = 0; fm < 2; ++fm) {
                acc[fm][fn] = __builtin_amdgcn_wmma_f32_16x16x32_bf16(
                    false, aHi[fm].v, false, bHi.v, (short)0, acc[fm][fn], false, false);
                acc[fm][fn] = __builtin_amdgcn_wmma_f32_16x16x32_bf16(
                    false, aHi[fm].v, false, bLo.v, (short)0, acc[fm][fn], false, false);
                acc[fm][fn] = __builtin_amdgcn_wmma_f32_16x16x32_bf16(
                    false, aLo[fm].v, false, bHi.v, (short)0, acc[fm][fn], false, false);
            }
        }
        __syncthreads();
    }

    // ---- epilogue ----
#pragma unroll
    for (int fm = 0; fm < 2; ++fm) {
#pragma unroll
        for (int fn = 0; fn < 4; ++fn) {
            int mbase = m0 + wm * 32 + fm * 16 + half * 8;
            int nn    = n0 + wn * 64 + fn * 16 + l16;
            float b = be[nn];
#pragma unroll
            for (int r = 0; r < 8; ++r) {
                float v = acc[fm][fn][r] + b;
                size_t off = eo * (size_t)M * N + (size_t)(mbase + r) * N + nn;
                if (FFN1) {
                    v = 0.5f * v * (1.0f + erff(v * 0.70710678118654752f));
                    unsigned short hi, lo;
                    split_bf16(v, hi, lo);
                    Chi[off] = hi;
                    Clo[off] = lo;
                } else {
                    Cf[off] = v;
                }
            }
        }
    }
}

// =====================================================================
// 5) Combine: out[t] = sum_k  p[t][k] * exp_out[e_k][slot_k]
// =====================================================================
__global__ void combine_kernel(const float* __restrict__ exp_out,  // [E][CAP][D]
                               const int* __restrict__ top_idx,
                               const float* __restrict__ top_p,
                               const int* __restrict__ combine_slot,
                               float* __restrict__ out)            // [NTOK][D]
{
    const int t = blockIdx.x;
    float4 a = make_float4(0.f, 0.f, 0.f, 0.f);
#pragma unroll
    for (int k = 0; k < 2; ++k) {
        int slot = combine_slot[t * 2 + k];
        if (slot >= 0) {
            int e   = top_idx[t * 2 + k];
            float w = top_p[t * 2 + k];
            float4 v = ((const float4*)exp_out)[((size_t)e * CAP + slot) * (DDIM / 4) + threadIdx.x];
            a.x += w * v.x; a.y += w * v.y; a.z += w * v.z; a.w += w * v.w;
        }
    }
    ((float4*)out)[(size_t)t * (DDIM / 4) + threadIdx.x] = a;
}

// =====================================================================
// launch
// =====================================================================
extern "C" void kernel_launch(void* const* d_in, const int* in_sizes, int n_in,
                              void* d_out, int out_size, void* d_ws, size_t ws_size,
                              hipStream_t stream) {
    const float* x         = (const float*)d_in[0];   // [4,1024,1024]
    const float* gate_w    = (const float*)d_in[1];   // [1024,8]
    const float* c_fc      = (const float*)d_in[2];   // [8,1024,4096]
    const float* fc_bias   = (const float*)d_in[3];   // [8,1,4096]
    const float* c_proj    = (const float*)d_in[4];   // [8,4096,1024]
    const float* proj_bias = (const float*)d_in[5];   // [8,1,1024]
    float* out = (float*)d_out;

    char* w = (char*)d_ws;
    size_t off = 0;
    int*   top_idx      = (int*)  (w + off); off += 32768;
    float* top_p        = (float*)(w + off); off += 32768;
    int*   combine_slot = (int*)  (w + off); off += 32768;
    int*   dispatch_tok = (int*)  (w + off); off += 40960;
    int*   used_count   = (int*)  (w + off); off += 256;
    unsigned short* xb_hi = (unsigned short*)(w + off); off += (size_t)NEXP*CAP*DDIM*2;
    unsigned short* xb_lo = (unsigned short*)(w + off); off += (size_t)NEXP*CAP*DDIM*2;
    unsigned short* h_hi  = (unsigned short*)(w + off); off += (size_t)NEXP*CAP*HDIM*2;
    unsigned short* h_lo  = (unsigned short*)(w + off); off += (size_t)NEXP*CAP*HDIM*2;
    float*          expo  = (float*)(w + off);          off += (size_t)NEXP*CAP*DDIM*4;

    router_kernel  <<<NTOK / 8, 256, 0, stream>>>(x, gate_w, top_idx, top_p);
    assign_kernel  <<<NEXP, 256, 0, stream>>>(top_idx, dispatch_tok, combine_slot, used_count);
    dispatch_kernel<<<dim3(CAP, NEXP), 256, 0, stream>>>(x, dispatch_tok, used_count, xb_hi, xb_lo);

    // FFN1: h = gelu(xb @ c_fc + fc_bias)   M=1280, N=4096, K=1024
    moe_gemm<true ><<<dim3(HDIM / 128, CAP / 128, NEXP), 256, 0, stream>>>(
        xb_hi, xb_lo, c_fc, fc_bias, h_hi, h_lo, (float*)nullptr, CAP, HDIM, DDIM);
    // FFN2: exp_out = h @ c_proj + proj_bias  M=1280, N=1024, K=4096
    moe_gemm<false><<<dim3(DDIM / 128, CAP / 128, NEXP), 256, 0, stream>>>(
        h_hi, h_lo, c_proj, proj_bias, (unsigned short*)nullptr, (unsigned short*)nullptr,
        expo, CAP, DDIM, HDIM);

    combine_kernel<<<NTOK, 256, 0, stream>>>(expo, top_idx, top_p, combine_slot, out);
}
